// Scaled_Dot_Product_Attention_70531952935216
// MI455X (gfx1250) — matrix operand order repaired and checked by reading, not compiled, not run on hardware
//
#include <hip/hip_runtime.h>
#include <hip/hip_bf16.h>

// ---------------------------------------------------------------------------
// Channel-attention (B=16, C=256, N=64*64=4096) for gfx1250 (MI455X, wave32)
// All matrix math on v_wmma_f32_16x16x32_f16 with fp32 accumulation.
// Round 2: software-pipelined fragment loads + wider per-wave tiles for
// higher WMMA-per-load ratio (reduce s_wait_loadcnt 0x0 serialization).
// ---------------------------------------------------------------------------

typedef __attribute__((ext_vector_type(16))) _Float16 v16h;
typedef __attribute__((ext_vector_type(8)))  _Float16 v8h;   // 16 bytes
typedef __attribute__((ext_vector_type(8)))  float    v8f;

#define BATCH 16
#define CCH   256
#define NPIX  4096

#define WMMA_F16(A, B, C) \
    __builtin_amdgcn_wmma_f32_16x16x32_f16(false, (A), false, (B), (short)0, (C), false, false)

union F16Frag { v16h v; v8h h[2]; };

// A-operand fragment (16x32, M x K), f16. Per CDNA5 ISA 7.12.2:
// lane L: M = L%16, h = L/16; chunk0 K = k0+8h..+7, chunk1 K = k0+16+8h..+7
__device__ inline v16h load_a_frag(const _Float16* base, int ldk, int m, int h, int k0) {
    F16Frag f;
    const _Float16* p = base + (size_t)m * ldk + k0 + 8 * h;
    f.h[0] = *(const v8h*)(p);
    f.h[1] = *(const v8h*)(p + 16);
    return f.v;
}

// B-operand fragment (32x16, K x N), f16. lane L: N = L%16, h = L/16;
// chunk0 K = k0+8h..+7, chunk1 K = k0+16+8h..+7 (same order as A).
__device__ inline v16h load_b_frag(const _Float16* base, int ldk, int n, int h, int k0) {
    F16Frag f;
    const _Float16* p = base + (size_t)n * ldk + k0 + 8 * h;
    f.h[0] = *(const v8h*)(p);
    f.h[1] = *(const v8h*)(p + 16);
    return f.v;
}

// ---------------------------------------------------------------------------
// Pass 0a: weights fp32 -> f16
// ---------------------------------------------------------------------------
__global__ __launch_bounds__(256) void convert_w_kernel(
    const float* __restrict__ wq, const float* __restrict__ wk, const float* __restrict__ wv,
    _Float16* __restrict__ WQ, _Float16* __restrict__ WK, _Float16* __restrict__ WV) {
    int i = blockIdx.x * 256 + threadIdx.x;
    WQ[i] = (_Float16)wq[i];
    WK[i] = (_Float16)wk[i];
    WV[i] = (_Float16)wv[i];
}

// ---------------------------------------------------------------------------
// Pass 0b: XT[b][n][ci] = (f16) x[b][ci][n]   (LDS-tiled 32x32 transpose)
// ---------------------------------------------------------------------------
__global__ __launch_bounds__(256) void convert_x_kernel(
    const float* __restrict__ x, _Float16* __restrict__ XT) {
    __shared__ _Float16 tile[32][33];
    const int b   = blockIdx.z;
    const int ci0 = blockIdx.y * 32;
    const int n0  = blockIdx.x * 32;
    const int tx = threadIdx.x;   // 0..31
    const int ty = threadIdx.y;   // 0..7
    const size_t xb = (size_t)b * CCH * NPIX;
#pragma unroll
    for (int j = 0; j < 4; ++j) {
        int ci = ci0 + ty + 8 * j;
        tile[ty + 8 * j][tx] = (_Float16)x[xb + (size_t)ci * NPIX + n0 + tx];
    }
    __syncthreads();
    const size_t tb = (size_t)b * NPIX * CCH;
#pragma unroll
    for (int j = 0; j < 4; ++j) {
        int n = n0 + ty + 8 * j;
        XT[tb + (size_t)n * CCH + ci0 + tx] = tile[tx][ty + 8 * j];
    }
}

// ---------------------------------------------------------------------------
// Pass 1: Q/K/V projections. One wave -> 16(co) x 32(n) tile of all three.
//   Per K-step: 2 X-fragments shared by 3 weight fragments -> 6 WMMA/10 loads.
//   Fragment loads for step k+1 issued before WMMAs of step k (pipelined).
// ---------------------------------------------------------------------------
__global__ __launch_bounds__(256) void proj_kernel(
    const _Float16* __restrict__ XT,
    const _Float16* __restrict__ WQ, const _Float16* __restrict__ WK, const _Float16* __restrict__ WV,
    const float* __restrict__ bq, const float* __restrict__ bk, const float* __restrict__ bv,
    _Float16* __restrict__ Q, _Float16* __restrict__ K, _Float16* __restrict__ VT) {
    const int w    = blockIdx.x * 8 + (threadIdx.x >> 5);
    const int lane = threadIdx.x & 31;
    const int l16  = lane & 15;
    const int h    = lane >> 4;
    const int b    = w >> 11;            // 2048 tiles per batch
    const int rem  = w & 2047;
    const int n0   = (rem >> 4) * 32;    // 128 n-tiles of width 32
    const int c0   = (rem & 15) * 16;    // 16 co-tiles

    const _Float16* xb = XT + (size_t)b * NPIX * CCH;

    v8f aq[2] = {}, ak[2] = {}, av[2] = {};
    v16h bx[2], fq, fk, fv;
#pragma unroll
    for (int j = 0; j < 2; ++j) bx[j] = load_b_frag(xb, CCH, n0 + 16 * j + l16, h, 0);
    fq = load_a_frag(WQ, CCH, c0 + l16, h, 0);
    fk = load_a_frag(WK, CCH, c0 + l16, h, 0);
    fv = load_a_frag(WV, CCH, c0 + l16, h, 0);

#pragma unroll
    for (int k0 = 0; k0 < CCH; k0 += 32) {
        v16h nbx[2], nfq, nfk, nfv;
        if (k0 + 32 < CCH) {             // prefetch next step's fragments
#pragma unroll
            for (int j = 0; j < 2; ++j)
                nbx[j] = load_b_frag(xb, CCH, n0 + 16 * j + l16, h, k0 + 32);
            nfq = load_a_frag(WQ, CCH, c0 + l16, h, k0 + 32);
            nfk = load_a_frag(WK, CCH, c0 + l16, h, k0 + 32);
            nfv = load_a_frag(WV, CCH, c0 + l16, h, k0 + 32);
        }
#pragma unroll
        for (int j = 0; j < 2; ++j) {
            aq[j] = WMMA_F16(fq, bx[j], aq[j]);
            ak[j] = WMMA_F16(fk, bx[j], ak[j]);
            av[j] = WMMA_F16(fv, bx[j], av[j]);
        }
        if (k0 + 32 < CCH) {
            bx[0] = nbx[0]; bx[1] = nbx[1];
            fq = nfq; fk = nfk; fv = nfv;
        }
    }

    // C/D layout: VGPR r holds (M = 8h + r, N = l16)
    const size_t qb = (size_t)b * CCH * NPIX;
#pragma unroll
    for (int j = 0; j < 2; ++j) {
        const int nj = n0 + 16 * j + l16;
#pragma unroll
        for (int r = 0; r < 8; ++r) {
            const int c = c0 + 8 * h + r;
            Q[qb + (size_t)c * NPIX + nj] = (_Float16)(aq[j][r] + bq[c]);
            K[qb + (size_t)c * NPIX + nj] = (_Float16)(ak[j][r] + bk[c]);
        }
        v8h vt;
#pragma unroll
        for (int r = 0; r < 8; ++r) vt[r] = (_Float16)(av[j][r] + bv[c0 + 8 * h + r]);
        *(v8h*)(VT + ((size_t)b * NPIX + nj) * CCH + c0 + 8 * h) = vt;  // 16B contig
    }
}

// ---------------------------------------------------------------------------
// Pass 2: logits L[b][cq][ck] = sum_n Q[cq][n]*K[ck][n].
//   One wave -> 32x32 output tile: 4 WMMA / 8 loads per 32-wide n-step,
//   pipelined across the 128-step reduction.
// ---------------------------------------------------------------------------
__global__ __launch_bounds__(256) void attn_kernel(
    const _Float16* __restrict__ Q, const _Float16* __restrict__ K, float* __restrict__ L) {
    const int w    = blockIdx.x * 8 + (threadIdx.x >> 5);
    const int lane = threadIdx.x & 31;
    const int l16  = lane & 15;
    const int h    = lane >> 4;
    const int b    = w >> 6;             // 64 tiles per batch (8x8 of 32x32)
    const int ci0  = ((w >> 3) & 7) * 32;
    const int cj0  = (w & 7) * 32;

    const _Float16* qb = Q + (size_t)b * CCH * NPIX;
    const _Float16* kb = K + (size_t)b * CCH * NPIX;

    v8f acc[2][2] = {};
    v16h af[2], bf[2];
#pragma unroll
    for (int i = 0; i < 2; ++i) {
        af[i] = load_a_frag(qb, NPIX, ci0 + 16 * i + l16, h, 0);
        bf[i] = load_b_frag(kb, NPIX, cj0 + 16 * i + l16, h, 0);
    }
    for (int n0 = 0; n0 < NPIX - 32; n0 += 32) {
        v16h naf[2], nbf[2];
#pragma unroll
        for (int i = 0; i < 2; ++i) {    // prefetch next n-step
            naf[i] = load_a_frag(qb, NPIX, ci0 + 16 * i + l16, h, n0 + 32);
            nbf[i] = load_b_frag(kb, NPIX, cj0 + 16 * i + l16, h, n0 + 32);
        }
#pragma unroll
        for (int i = 0; i < 2; ++i)
#pragma unroll
            for (int j = 0; j < 2; ++j)
                acc[i][j] = WMMA_F16(af[i], bf[j], acc[i][j]);
#pragma unroll
        for (int i = 0; i < 2; ++i) { af[i] = naf[i]; bf[i] = nbf[i]; }
    }
#pragma unroll
    for (int i = 0; i < 2; ++i)          // final (drain) step
#pragma unroll
        for (int j = 0; j < 2; ++j)
            acc[i][j] = WMMA_F16(af[i], bf[j], acc[i][j]);

    const size_t base = (size_t)b * CCH;
#pragma unroll
    for (int i = 0; i < 2; ++i)
#pragma unroll
        for (int j = 0; j < 2; ++j)
#pragma unroll
            for (int r = 0; r < 8; ++r)
                L[(base + ci0 + 16 * i + 8 * h + r) * CCH + cj0 + 16 * j + l16] = acc[i][j][r];
}

// ---------------------------------------------------------------------------
// Pass 3: row softmax over 256 logits (scale 1/16), emit f16 probs
// ---------------------------------------------------------------------------
__global__ __launch_bounds__(256) void softmax_kernel(
    const float* __restrict__ L, _Float16* __restrict__ P) {
    __shared__ float red[256];
    const int row = blockIdx.x;
    const int t   = threadIdx.x;
    const float v = L[(size_t)row * CCH + t] * 0.0625f;
    red[t] = v;
    __syncthreads();
    for (int s = 128; s > 0; s >>= 1) {
        if (t < s) red[t] = fmaxf(red[t], red[t + s]);
        __syncthreads();
    }
    const float m = red[0];
    __syncthreads();
    const float e = __expf(v - m);
    red[t] = e;
    __syncthreads();
    for (int s = 128; s > 0; s >>= 1) {
        if (t < s) red[t] += red[t + s];
        __syncthreads();
    }
    P[(size_t)row * CCH + t] = (_Float16)(e / red[0]);
}

// ---------------------------------------------------------------------------
// Pass 4: out = x + P @ V. One wave -> 16(c) x 64(n): one P fragment feeds
//   4 V^T fragments -> 4 WMMA / 10 loads per d-step (8 steps), pipelined.
// ---------------------------------------------------------------------------
__global__ __launch_bounds__(256) void attended_kernel(
    const _Float16* __restrict__ P, const _Float16* __restrict__ VT,
    const float* __restrict__ x, float* __restrict__ out) {
    const int w    = blockIdx.x * 8 + (threadIdx.x >> 5);
    const int lane = threadIdx.x & 31;
    const int l16  = lane & 15;
    const int h    = lane >> 4;
    const int b    = w >> 10;            // 1024 tiles per batch
    const int rem  = w & 1023;
    const int n0   = (rem >> 4) * 64;    // 64 n-tiles of width 64
    const int c0   = (rem & 15) * 16;    // 16 c-tiles

    const _Float16* pb = P  + (size_t)b * CCH * CCH;
    const _Float16* vb = VT + (size_t)b * NPIX * CCH;

    v8f acc[4] = {};
    v16h af, bf[4];
    af = load_a_frag(pb, CCH, c0 + l16, h, 0);
#pragma unroll
    for (int j = 0; j < 4; ++j) bf[j] = load_b_frag(vb, CCH, n0 + 16 * j + l16, h, 0);

#pragma unroll
    for (int d0 = 0; d0 < CCH; d0 += 32) {
        v16h naf, nbf[4];
        if (d0 + 32 < CCH) {             // prefetch next d-step
            naf = load_a_frag(pb, CCH, c0 + l16, h, d0 + 32);
#pragma unroll
            for (int j = 0; j < 4; ++j)
                nbf[j] = load_b_frag(vb, CCH, n0 + 16 * j + l16, h, d0 + 32);
        }
#pragma unroll
        for (int j = 0; j < 4; ++j)
            acc[j] = WMMA_F16(af, bf[j], acc[j]);
        if (d0 + 32 < CCH) {
            af = naf;
#pragma unroll
            for (int j = 0; j < 4; ++j) bf[j] = nbf[j];
        }
    }

    const size_t xb = (size_t)b * CCH * NPIX;
#pragma unroll
    for (int j = 0; j < 4; ++j) {
        const int nj = n0 + 16 * j + l16;
#pragma unroll
        for (int r = 0; r < 8; ++r) {
            const int c = c0 + 8 * h + r;
            const size_t idx = xb + (size_t)c * NPIX + nj;
            out[idx] = x[idx] + acc[j][r];
        }
    }
}

// ---------------------------------------------------------------------------
extern "C" void kernel_launch(void* const* d_in, const int* in_sizes, int n_in,
                              void* d_out, int out_size, void* d_ws, size_t ws_size,
                              hipStream_t stream) {
    const float* x  = (const float*)d_in[0];
    const float* wq = (const float*)d_in[1];
    const float* bq = (const float*)d_in[2];
    const float* wk = (const float*)d_in[3];
    const float* bk = (const float*)d_in[4];
    const float* wv = (const float*)d_in[5];
    const float* bv = (const float*)d_in[6];
    float* out = (float*)d_out;

    // Workspace carve-up (halves unless noted). Total ~134.4 MiB.
    const size_t XN = (size_t)BATCH * NPIX * CCH;   // 16.78M elements
    _Float16* XT  = (_Float16*)d_ws;
    _Float16* WQh = XT + XN;
    _Float16* WKh = WQh + (size_t)CCH * CCH;
    _Float16* WVh = WKh + (size_t)CCH * CCH;
    _Float16* Qh  = WVh + (size_t)CCH * CCH;
    _Float16* Kh  = Qh + XN;
    _Float16* VTh = Kh + XN;
    float*    attnL = (float*)(VTh + XN);                        // [B,256,256] fp32
    _Float16* attnP = (_Float16*)(attnL + (size_t)BATCH * CCH * CCH);

    convert_w_kernel<<<(CCH * CCH) / 256, 256, 0, stream>>>(wq, wk, wv, WQh, WKh, WVh);
    convert_x_kernel<<<dim3(NPIX / 32, CCH / 32, BATCH), dim3(32, 8), 0, stream>>>(x, XT);

    // Pass 1: 16co x 32n wave tiles -> 32768 waves / 4096 blocks
    proj_kernel<<<(BATCH * (NPIX / 32) * (CCH / 16)) / 8, 256, 0, stream>>>(
        XT, WQh, WKh, WVh, bq, bk, bv, Qh, Kh, VTh);

    // Pass 2: 32x32 wave tiles -> 1024 waves / 128 blocks
    attn_kernel<<<(BATCH * (CCH / 32) * (CCH / 32)) / 8, 256, 0, stream>>>(Qh, Kh, attnL);

    // Pass 3: softmax, one 256-thread block per row
    softmax_kernel<<<BATCH * CCH, 256, 0, stream>>>(attnL, attnP);

    // Pass 4: 16c x 64n wave tiles -> 16384 waves / 2048 blocks
    attended_kernel<<<(BATCH * (CCH / 16) * (NPIX / 64)) / 8, 256, 0, stream>>>(
        attnP, VTh, x, out);
}
